// CognitiveOrganism_32899449488193
// MI455X (gfx1250) — compile-verified, matrix operand order checked by reading, not run on hardware
//
#include <hip/hip_runtime.h>
#include <hip/hip_bf16.h>
#include <cstdint>

// CDNA5 / gfx1250, wave32. D = A(16x32 bf16) x B(32x16 bf16) + C(16x16 f32)
typedef __attribute__((ext_vector_type(16))) __bf16 v16bf;
typedef __attribute__((ext_vector_type(8)))  float  v8f;

// ---- packed weight-tile layout in d_ws (each tile = 32 lanes x 16 bf16 = 1KB) ----
#define OFF_WBIT  0    // 2 tiles : W_bit^T   (K=32 pad from 8, N-tiles 0..1)
#define OFF_WBR   2    // 4 tiles : W_bridge^T (K=32, N-tiles 0..3)
#define OFF_WSH   6    // 8 tiles : W_sh^T    (kt*4+nt)
#define OFF_WHALT 14   // 4 tiles : halt_w    (kt*2+nt), n = l*8+r
#define OFF_WREAD 18   // 16 tiles: W_read^T  (K=512, N=8 pad 16), for correction GEMM
#define OFF_BBIG  34   // 64 tiles: decay_g*W_read/L fused (K=2048, N=8 pad 16)
#define NTILES    98

static __device__ __forceinline__ __bf16 tobf(float f) {
  return (__bf16)f;   // native RTNE convert (v_cvt_*bf16_f32 on gfx1250)
}
static __device__ __forceinline__ float sigm(float x) {
  return 1.0f / (1.0f + __expf(-x));
}
static __device__ __forceinline__ v8f v8f_zero() {
  v8f c = {0.f, 0.f, 0.f, 0.f, 0.f, 0.f, 0.f, 0.f};
  return c;
}
static __device__ __forceinline__ v8f wmma_bf16(v16bf a, v16bf b, v8f c) {
  // 8 args: (neg_a, A, neg_b, B, c_mod, C, reuse_a, reuse_b)
  return __builtin_amdgcn_wmma_f32_16x16x32_bf16(false, a, false, b, (short)0, c,
                                                 false, false);
}
static __device__ __forceinline__ v16bf load_btile(const __bf16* wsp, int tile, int lane) {
  return ((const v16bf*)(wsp + tile * 512))[lane];
}
// A-matrix lane layout (16-bit 16x32): lane holds row m=lane&15; K elems:
// lanes 0-15 : e0..7 -> K=kb+0..7,   e8..15 -> K=kb+16..23
// lanes 16-31: e0..7 -> K=kb+8..15,  e8..15 -> K=kb+24..31   (kh = (lane>>4)*8)
static __device__ __forceinline__ v16bf build_a(const float* row, int kbase, int kh) {
  v16bf a;
#pragma unroll
  for (int e = 0; e < 8; ++e) a[e] = tobf(row[kbase + kh + e]);
#pragma unroll
  for (int e = 0; e < 8; ++e) a[8 + e] = tobf(row[kbase + kh + 16 + e]);
  return a;
}

// ---------------- prep: pack all B-matrices into per-lane WMMA layout -------------
__global__ void prep_kernel(const float* __restrict__ W_bit,
                            const float* __restrict__ W_bridge,
                            const float* __restrict__ W_sh,
                            const float* __restrict__ halt_w,
                            const float* __restrict__ W_read,
                            const float* __restrict__ decays,
                            unsigned short* __restrict__ wsb) {
  int idx = blockIdx.x * blockDim.x + threadIdx.x;
  if (idx >= NTILES * 512) return;
  int tile = idx >> 9;
  int within = idx & 511;
  int lane = within >> 4;
  int e = within & 15;
  int n = lane & 15;                                  // B col (N) within tile
  int k = (e < 8 ? e : e + 8) + ((lane >> 4) << 3);   // B row (K) within tile
  float val = 0.0f;
  if (tile < OFF_WBR) {
    int nt = tile - OFF_WBIT;
    if (k < 8) val = W_bit[(nt * 16 + n) * 8 + k];
  } else if (tile < OFF_WSH) {
    int nt = tile - OFF_WBR;
    val = W_bridge[(nt * 16 + n) * 32 + k];
  } else if (tile < OFF_WHALT) {
    int t = tile - OFF_WSH;
    int kt = t >> 2, nt = t & 3;
    val = W_sh[(nt * 16 + n) * 64 + kt * 32 + k];
  } else if (tile < OFF_WREAD) {
    int t = tile - OFF_WHALT;
    int kt = t >> 1, nt = t & 1;
    int nf = nt * 16 + n;                             // nf = l*8+r
    val = halt_w[nf * 64 + kt * 32 + k];
  } else if (tile < OFF_BBIG) {
    int kt = tile - OFF_WREAD;
    if (n < 8) val = W_read[n * 512 + kt * 32 + k];
  } else {
    int kt = tile - OFF_BBIG;
    int K = kt * 32 + k;                              // flat (l, r*64+m)
    int l = K >> 9, rm = K & 511;
    if (n < 8) {
      float dg = sigm(decays[l * 512 + rm]);
      val = dg * 0.25f * W_read[n * 512 + rm];        // 0.25 = mean over L
    }
  }
  wsb[idx] = __builtin_bit_cast(unsigned short, tobf(val));
}

// ---------------- main: 4 waves/block, 16 batch rows per wave ---------------------
__global__ __launch_bounds__(128) void organism_kernel(
    const float* __restrict__ x, const float* __restrict__ H,
    const float* __restrict__ b_bit, const float* __restrict__ b_read,
    const float* __restrict__ b_sh,
    const float* __restrict__ W_eng, const float* __restrict__ b_eng,
    const float* __restrict__ W_imp, const float* __restrict__ b_imp,
    const float* __restrict__ W_sur, const float* __restrict__ b_sur,
    const float* __restrict__ W_gate, const float* __restrict__ b_gate,
    const float* __restrict__ halt_b,
    const unsigned short* __restrict__ wsb, float* __restrict__ out) {
  __shared__ float bufA[4][16][64];   // s1 -> hg -> halt (per wave)
  __shared__ float bufB[4][16][64];   // s2 (per wave)
  __shared__ float aux[4][16][16];    // [0..7]=cSHs[r], [8]=surprise, [9]=eng*imp
  const __bf16* wsp = (const __bf16*)wsb;

  const int lane = threadIdx.x & 31;
  const int w = threadIdx.x >> 5;
  const int b0 = (blockIdx.x * 4 + w) * 16;     // requires B % 64 == 0 (B=65536)
  const int m = lane & 15;                      // A-row / batch row within tile
  const int kh = (lane >> 4) << 3;              // K sub-offset per lane half
  const int n = m;                              // D-col for this lane
  const int mb = kh;                            // D rows: vgpr i -> row mb+i

  // ---- s1 = x @ W_bit^T + b_bit  (K padded 8 -> 32) ----
  v16bf ax;
  {
    __bf16 z = __builtin_bit_cast(__bf16, (unsigned short)0);
#pragma unroll
    for (int e = 0; e < 16; ++e) ax[e] = z;
  }
  if (lane < 16) {  // only K=0..7 are real; lanes>=16 hold K=8..15,24..31 (all zero)
    const float4 x0 = *(const float4*)(x + (size_t)(b0 + m) * 8);
    const float4 x1 = *(const float4*)(x + (size_t)(b0 + m) * 8 + 4);
    ax[0] = tobf(x0.x); ax[1] = tobf(x0.y); ax[2] = tobf(x0.z); ax[3] = tobf(x0.w);
    ax[4] = tobf(x1.x); ax[5] = tobf(x1.y); ax[6] = tobf(x1.z); ax[7] = tobf(x1.w);
  }
#pragma unroll
  for (int nt = 0; nt < 2; ++nt) {
    v8f c = v8f_zero();
    c = wmma_bf16(ax, load_btile(wsp, OFF_WBIT + nt, lane), c);
    float bb = b_bit[nt * 16 + n];
#pragma unroll
    for (int i = 0; i < 8; ++i) bufA[w][mb + i][nt * 16 + n] = c[i] + bb;
  }
  __syncthreads();

  // ---- surprise = sig(s1.Wsur)*sig(s1.Wgate) ----
  if (lane < 16) {
    float s = 0.f, g = 0.f;
    const float* r1 = &bufA[w][lane][0];
#pragma unroll 8
    for (int k = 0; k < 32; ++k) { float v = r1[k]; s += v * W_sur[k]; g += v * W_gate[k]; }
    aux[w][lane][8] = sigm(s + b_sur[0]) * sigm(g + b_gate[0]);
  }

  // ---- s2 = tanh(s1 @ W_bridge^T) ----
  {
    const float* rowS1 = &bufA[w][m][0];
    v16bf a = build_a(rowS1, 0, kh);
#pragma unroll
    for (int nt = 0; nt < 4; ++nt) {
      v8f c = v8f_zero();
      c = wmma_bf16(a, load_btile(wsp, OFF_WBR + nt, lane), c);
#pragma unroll
      for (int i = 0; i < 8; ++i) bufB[w][mb + i][nt * 16 + n] = tanhf(c[i]);
    }
  }
  __syncthreads();

  // ---- hg = relu(s2 @ W_sh^T + b_sh) ----
  {
    const float* rowS2 = &bufB[w][m][0];
    v16bf a0 = build_a(rowS2, 0, kh);
    v16bf a1 = build_a(rowS2, 32, kh);
#pragma unroll
    for (int nt = 0; nt < 4; ++nt) {
      v8f c = v8f_zero();
      c = wmma_bf16(a0, load_btile(wsp, OFF_WSH + nt, lane), c);
      c = wmma_bf16(a1, load_btile(wsp, OFF_WSH + 4 + nt, lane), c);
      float bb = b_sh[nt * 16 + n];
#pragma unroll
      for (int i = 0; i < 8; ++i) bufA[w][mb + i][nt * 16 + n] = fmaxf(c[i] + bb, 0.f);
    }
  }
  __syncthreads();

  // ---- eng*imp ----
  if (lane < 16) {
    float e1 = 0.f, e2 = 0.f;
    const float* rh = &bufA[w][lane][0];
#pragma unroll 8
    for (int k = 0; k < 64; ++k) { float v = rh[k]; e1 += v * W_eng[k]; e2 += v * W_imp[k]; }
    aux[w][lane][9] = sigm(e1 + b_eng[0]) * sigm(e2 + b_imp[0]);
  }
  __syncthreads();

  // ---- halt[l,r] = sig(s2 . halt_w[l,r] + halt_b[l]) ; n = l*8+r ----
  {
    const float* rowS2 = &bufB[w][m][0];
    v16bf a0 = build_a(rowS2, 0, kh);
    v16bf a1 = build_a(rowS2, 32, kh);
#pragma unroll
    for (int nt = 0; nt < 2; ++nt) {
      v8f c = v8f_zero();
      c = wmma_bf16(a0, load_btile(wsp, OFF_WHALT + nt, lane), c);
      c = wmma_bf16(a1, load_btile(wsp, OFF_WHALT + 2 + nt, lane), c);
      int nf = nt * 16 + n;
      float hb = halt_b[nf >> 3];
#pragma unroll
      for (int i = 0; i < 8; ++i) bufA[w][mb + i][nf] = sigm(c[i] + hb);
    }
  }
  __syncthreads();

  // ---- cSHs[r] = surprise*eng*imp/L * sum_l halt[l,r] ----
  if (lane < 16) {
    float csc = aux[w][lane][8] * aux[w][lane][9] * 0.25f;
#pragma unroll
    for (int r = 0; r < 8; ++r) {
      float sh = bufA[w][lane][r] + bufA[w][lane][8 + r] +
                 bufA[w][lane][16 + r] + bufA[w][lane][24 + r];
      aux[w][lane][r] = csc * sh;
    }
  }
  __syncthreads();

  // ---- main HBM-bound pass: out = H_flat @ Bbig  (+ rank-1 correction GEMM) ----
  v8f acc = v8f_zero();
  const float* Hrow = H + (size_t)(b0 + m) * 2048;
#pragma unroll 2
  for (int kt = 0; kt < 64; ++kt) {
    int kb = kt * 32 + kh;
    // pull the stream ~1KB ahead (speculative; overshoot past end is dropped)
    __builtin_prefetch(Hrow + kb + 256, 0, 1);
    float4 h0 = *(const float4*)(Hrow + kb);
    float4 h1 = *(const float4*)(Hrow + kb + 4);
    float4 h2 = *(const float4*)(Hrow + kb + 16);
    float4 h3 = *(const float4*)(Hrow + kb + 20);
    v16bf a;
    a[0]  = tobf(h0.x); a[1]  = tobf(h0.y); a[2]  = tobf(h0.z); a[3]  = tobf(h0.w);
    a[4]  = tobf(h1.x); a[5]  = tobf(h1.y); a[6]  = tobf(h1.z); a[7]  = tobf(h1.w);
    a[8]  = tobf(h2.x); a[9]  = tobf(h2.y); a[10] = tobf(h2.z); a[11] = tobf(h2.w);
    a[12] = tobf(h3.x); a[13] = tobf(h3.y); a[14] = tobf(h3.z); a[15] = tobf(h3.w);
    acc = wmma_bf16(a, load_btile(wsp, OFF_BBIG + kt, lane), acc);
  }
  {
    // V[b, r*64+mm] = cSHs[b,r] * s2[b,mm], GEMM against packed W_read^T (K=512)
    const float* rowS2 = &bufB[w][m][0];
    const float* csh = &aux[w][m][0];
#pragma unroll 2
    for (int kt = 0; kt < 16; ++kt) {
      int kb = kt * 32 + kh;
      v16bf a;
      {
        float cs = csh[kb >> 6]; int mm = kb & 63;
#pragma unroll
        for (int e = 0; e < 8; ++e) a[e] = tobf(cs * rowS2[mm + e]);
      }
      {
        int k2 = kb + 16;
        float cs = csh[k2 >> 6]; int mm = k2 & 63;
#pragma unroll
        for (int e = 0; e < 8; ++e) a[8 + e] = tobf(cs * rowS2[mm + e]);
      }
      acc = wmma_bf16(a, load_btile(wsp, OFF_WREAD + kt, lane), acc);
    }
  }
  // ---- bias + store: lane n (<8) holds column o=n for rows mb..mb+7 ----
  if (n < 8) {
    float br = b_read[n];
#pragma unroll
    for (int i = 0; i < 8; ++i)
      out[(size_t)(b0 + mb + i) * 8 + n] = acc[i] + br;
  }
}

extern "C" void kernel_launch(void* const* d_in, const int* in_sizes, int n_in,
                              void* d_out, int out_size, void* d_ws, size_t ws_size,
                              hipStream_t stream) {
  const float* x        = (const float*)d_in[0];
  const float* H        = (const float*)d_in[1];
  const float* W_bit    = (const float*)d_in[2];
  const float* b_bit    = (const float*)d_in[3];
  const float* W_bridge = (const float*)d_in[4];
  const float* W_read   = (const float*)d_in[5];
  const float* b_read   = (const float*)d_in[6];
  const float* W_sh     = (const float*)d_in[7];
  const float* b_sh     = (const float*)d_in[8];
  const float* W_eng    = (const float*)d_in[9];
  const float* b_eng    = (const float*)d_in[10];
  const float* W_imp    = (const float*)d_in[11];
  const float* b_imp    = (const float*)d_in[12];
  const float* W_sur    = (const float*)d_in[13];
  const float* b_sur    = (const float*)d_in[14];
  const float* W_gate   = (const float*)d_in[15];
  const float* b_gate   = (const float*)d_in[16];
  const float* decays   = (const float*)d_in[17];
  const float* halt_w   = (const float*)d_in[18];
  const float* halt_b   = (const float*)d_in[19];
  float* out = (float*)d_out;
  unsigned short* wsb = (unsigned short*)d_ws;  // 98 KB of packed bf16 weight tiles

  int B = in_sizes[0] / 8;  // 65536; kernel assumes B % 64 == 0

  hipLaunchKernelGGL(prep_kernel, dim3((NTILES * 512 + 255) / 256), dim3(256), 0,
                     stream, W_bit, W_bridge, W_sh, halt_w, W_read, decays, wsb);
  hipLaunchKernelGGL(organism_kernel, dim3(B / 64), dim3(128), 0, stream,
                     x, H, b_bit, b_read, b_sh, W_eng, b_eng, W_imp, b_imp,
                     W_sur, b_sur, W_gate, b_gate, halt_b, wsb, out);
}
